// Encoder_MF_21603685499265
// MI455X (gfx1250) — matrix operand order repaired and checked by reading, not compiled, not run on hardware
//
#include <hip/hip_runtime.h>
#include <hip/hip_bf16.h>
#include <math.h>

// ---------------------------------------------------------------------------
// Attention-LSTM encoder for MI455X (gfx1250, wave32, WMMA).
// B=256, D=128, F=8, T=128, H=128.  K_ih = D*F = 1024, NG = 4H = 512.
//
// Batch-parallel (16 batch tiles of M=16), sequential T loop inside a
// persistent workgroup (one per WGP).  Gates GEMM per step per workgroup:
//   (16 x 1024) @ (1024 x 512)  +  (16 x 128) @ (128 x 512)
// via v_wmma_f32_16x16x32_f16, f32 accumulate seeded with the bias.
// Weights pre-packed (one-time kernel) into per-lane B-fragment order so a
// fragment is a contiguous 32B load; x_hat/h written to LDS in A-fragment
// order so a fragment is a contiguous 32B ds read.  W_hh fragments staged
// into LDS once (reused 128x).  Fragment fetches are manually double-
// buffered so s_wait_* overlaps the WMMA pipeline.
// ---------------------------------------------------------------------------

typedef _Float16 v16h __attribute__((ext_vector_type(16)));
typedef _Float16 v8h  __attribute__((ext_vector_type(8)));
typedef float    v8f  __attribute__((ext_vector_type(8)));

#define B_   256
#define D_   128
#define F_   8
#define T_   128
#define H_   128
#define KIH  1024   // D*F
#define NG   512    // 4*H

// workspace element counts
#define WIH_HALVES (32 * 32 * 32 * 16)  // 524288 : [ntile32][ktile32][lane32][16]
#define WHH_HALVES (32 * 4 * 32 * 16)   // 65536  : [ntile32][ktile4][lane32][16]
#define EX_FLOATS  (B_ * D_)            // 32768
#define BIAS_FLOATS NG                  // 512

__device__ __forceinline__ float sigf(float x) {
    return 1.0f / (1.0f + __expf(-x));
}
__device__ __forceinline__ float tanhg(float x) {
    return 2.0f / (1.0f + __expf(-2.0f * x)) - 1.0f;
}

// ---------------------------------------------------------------------------
// Prep kernel 1: pack W_ih (512x1024) and W_hh (512x128) into f16 B-fragment
// order, and fuse bias.  B-matrix (KxN = W^T) per-lane layout for 16-bit
// 32x16 tiles:  lane l -> n = l&15, khalf = l>>4 ; half j -> k = khalf*8 + (j&7) + (j>>3)*16.
// ---------------------------------------------------------------------------
__global__ void enc_pack_kernel(const float* __restrict__ W_ih,
                                const float* __restrict__ W_hh,
                                const float* __restrict__ b_ih,
                                const float* __restrict__ b_hh,
                                _Float16* __restrict__ wih,
                                _Float16* __restrict__ whh,
                                float* __restrict__ biasw) {
    int id = blockIdx.x * blockDim.x + threadIdx.x;
    if (id < WIH_HALVES) {
        int j  = id & 15;
        int ln = (id >> 4) & 31;
        int kt = (id >> 9) & 31;
        int nt = id >> 14;
        int n = nt * 16 + (ln & 15);
        int k = kt * 32 + (ln >> 4) * 8 + (j & 7) + (j >> 3) * 16;
        wih[id] = (_Float16)W_ih[n * KIH + k];
    } else if (id < WIH_HALVES + WHH_HALVES) {
        int q  = id - WIH_HALVES;
        int j  = q & 15;
        int ln = (q >> 4) & 31;
        int kt = (q >> 9) & 3;
        int nt = q >> 11;
        int n = nt * 16 + (ln & 15);
        int k = kt * 32 + (ln >> 4) * 8 + (j & 7) + (j >> 3) * 16;
        whh[q] = (_Float16)W_hh[n * H_ + k];
    } else if (id < WIH_HALVES + WHH_HALVES + BIAS_FLOATS) {
        int n = id - (WIH_HALVES + WHH_HALVES);
        biasw[n] = b_ih[n] + b_hh[n];
    }
}

// ---------------------------------------------------------------------------
// Prep kernel 2: ex[b,d] = sum_k x_flat[b,d,k] * w_x[k]  (k = f*T + t, 1024)
// one wave per (b,d) dot product, coalesced, shfl reduction.
// ---------------------------------------------------------------------------
__global__ void enc_ex_kernel(const float* __restrict__ input,
                              const float* __restrict__ w_attn,
                              float* __restrict__ exw) {
    int gid   = blockIdx.x * blockDim.x + threadIdx.x;
    int gwave = gid >> 5;
    int lane  = gid & 31;
    if (gwave >= B_ * D_) return;
    const float* xp = input + (size_t)gwave * KIH;
    const float* wx = w_attn + 2 * H_;
    float s = 0.0f;
#pragma unroll 8
    for (int i = 0; i < 32; ++i) {
        int k = lane + 32 * i;
        s += xp[k] * wx[k];
    }
    for (int off = 16; off; off >>= 1) s += __shfl_xor(s, off, 32);
    if (lane == 0) exw[gwave] = s;
}

// ---------------------------------------------------------------------------
// Main persistent kernel: 16 blocks x 512 threads (16 wave32 per block).
// Block = one batch tile of 16.  Wave w: attention row m=w AND gates
// N-chunk [32w, 32w+32).
// ---------------------------------------------------------------------------
__global__ void __launch_bounds__(512)
enc_main_kernel(const float* __restrict__ input,
                const float* __restrict__ w_attn,
                const float* __restrict__ b_attn,
                const _Float16* __restrict__ wih,
                const _Float16* __restrict__ whh,
                const float* __restrict__ exw,
                const float* __restrict__ biasw,
                float* __restrict__ out) {
    // LDS (~221 KB of 320 KB per WGP; one workgroup per WGP by design)
    __shared__ __attribute__((aligned(32))) _Float16 lds_whh[WHH_HALVES];   // 128KB staged W_hh B-fragments
    __shared__ __attribute__((aligned(32))) _Float16 lds_x[32 * 32 * 16];   // 32KB x_hat A-fragments [kt][lane][16]
    __shared__ __attribute__((aligned(32))) _Float16 lds_h[4 * 32 * 16];    // 4KB  h A-fragments [kt][lane][16]
    __shared__ float lds_gates[16 * NG];   // 32KB [m][n]
    __shared__ float lds_hf[16 * H_];      // 8KB  h f32 [m][j]
    __shared__ float lds_cf[16 * H_];      // 8KB  c f32 [m][j]
    __shared__ float lds_ex[16 * D_];      // 8KB  ex tile [m][d]
    __shared__ float lds_wh[H_];
    __shared__ float lds_wc[H_];

    const int tid   = threadIdx.x;
    const int wave  = tid >> 5;
    const int lane  = tid & 31;
    const int btile = blockIdx.x;            // 0..15
    const int m     = wave;                  // batch row handled for attention
    const int bglob = btile * 16 + m;

    // ---- one-time init ----
    for (int i = tid; i < 16 * H_; i += 512) {
        lds_hf[i] = 0.0f;
        lds_cf[i] = 0.0f;
    }
    for (int i = tid; i < 16 * D_; i += 512) {
        int mm = i >> 7, d = i & 127;
        lds_ex[i] = exw[(size_t)(btile * 16 + mm) * D_ + d];
    }
    for (int i = tid; i < 4 * 32 * 16; i += 512) lds_h[i] = (_Float16)0.0f;
    // stage all W_hh B-fragments into LDS (reused every step)
    for (int i = tid; i < WHH_HALVES / 8; i += 512) {
        *(v8h*)&lds_whh[i * 8] = *(const v8h*)&whh[(size_t)i * 8];
    }
    if (tid < H_) {
        lds_wh[tid] = w_attn[tid];
        lds_wc[tid] = w_attn[H_ + tid];
    }
    const float battn = b_attn[0];
    const float bias0 = biasw[wave * 32 + (lane & 15)];
    const float bias1 = biasw[wave * 32 + 16 + (lane & 15)];
    __syncthreads();

    const int nt0 = 2 * wave, nt1 = 2 * wave + 1;

    for (int t = 0; t < T_; ++t) {
        // ---- stage 1: attention scalar + softmax + x_hat (wave w = batch m) ----
        float part = 0.0f;
#pragma unroll
        for (int q = 0; q < 4; ++q) {
            int d = q * 32 + lane;
            part += lds_hf[m * H_ + d] * lds_wh[d] + lds_cf[m * H_ + d] * lds_wc[d];
        }
        for (int off = 16; off; off >>= 1) part += __shfl_xor(part, off, 32);
        const float s = part + battn;

        float e[4];
        float mx = -3.4e38f;
#pragma unroll
        for (int q = 0; q < 4; ++q) {
            int d = q * 32 + lane;
            e[q] = lds_ex[m * D_ + d] + s;
            mx = fmaxf(mx, e[q]);
        }
        for (int off = 16; off; off >>= 1) mx = fmaxf(mx, __shfl_xor(mx, off, 32));
        float sum = 0.0f;
#pragma unroll
        for (int q = 0; q < 4; ++q) {
            e[q] = __expf(e[q] - mx);
            sum += e[q];
        }
        for (int off = 16; off; off >>= 1) sum += __shfl_xor(sum, off, 32);
        const float inv = 1.0f / sum;

#pragma unroll
        for (int q = 0; q < 4; ++q) {
            int d = q * 32 + lane;
            float a = e[q] * inv;
            // input[b][d][f][t] = (b*128+d)*1024 + f*128 + t
            const float* xp = input + ((size_t)bglob * D_ + d) * (F_ * T_) + t;
            v8h hx;
#pragma unroll
            for (int f = 0; f < F_; ++f) hx[f] = (_Float16)(a * xp[f * T_]);
            // A-fragment pack: k = d*8 .. d*8+7  (contiguous 16B in one lane)
            int kt = d >> 2;
            int ln = (d & 1) ? (16 + m) : m;
            int jj = (d & 2) ? 8 : 0;
            *(v8h*)&lds_x[(kt * 32 + ln) * 16 + jj] = hx;
        }
        __syncthreads();

        // ---- stage 2: gates = W_ih@x_hat + W_hh@h + bias via WMMA ----
        v8f c0, c1;
#pragma unroll
        for (int r = 0; r < 8; ++r) { c0[r] = bias0; c1[r] = bias1; }

        // W_hh (K = 128) from LDS, double-buffered
        {
            v16h ah  = *(const v16h*)&lds_h[lane * 16];
            v16h bh0 = *(const v16h*)&lds_whh[((nt0 * 4) * 32 + lane) * 16];
            v16h bh1 = *(const v16h*)&lds_whh[((nt1 * 4) * 32 + lane) * 16];
#pragma unroll
            for (int kt = 0; kt < 4; ++kt) {
                int kn = (kt + 1) & 3;
                v16h an  = *(const v16h*)&lds_h[(kn * 32 + lane) * 16];
                v16h bn0 = *(const v16h*)&lds_whh[((nt0 * 4 + kn) * 32 + lane) * 16];
                v16h bn1 = *(const v16h*)&lds_whh[((nt1 * 4 + kn) * 32 + lane) * 16];
                c0 = __builtin_amdgcn_wmma_f32_16x16x32_f16(false, ah, false, bh0,
                                                            (short)0, c0, false, false);
                c1 = __builtin_amdgcn_wmma_f32_16x16x32_f16(false, ah, false, bh1,
                                                            (short)0, c1, false, false);
                ah = an; bh0 = bn0; bh1 = bn1;
            }
        }
        // W_ih (K = 1024), A from LDS, B streamed from L2-resident packed
        // weights; next tile's fragments fetched before current WMMAs issue.
        {
            v16h ah  = *(const v16h*)&lds_x[lane * 16];
            v16h bh0 = *(const v16h*)&wih[((size_t)(nt0 * 32) * 32 + lane) * 16];
            v16h bh1 = *(const v16h*)&wih[((size_t)(nt1 * 32) * 32 + lane) * 16];
#pragma unroll 4
            for (int kt = 0; kt < 32; ++kt) {
                int kn = (kt + 1) & 31;
                v16h an  = *(const v16h*)&lds_x[(kn * 32 + lane) * 16];
                v16h bn0 = *(const v16h*)&wih[((size_t)(nt0 * 32 + kn) * 32 + lane) * 16];
                v16h bn1 = *(const v16h*)&wih[((size_t)(nt1 * 32 + kn) * 32 + lane) * 16];
                c0 = __builtin_amdgcn_wmma_f32_16x16x32_f16(false, ah, false, bh0,
                                                            (short)0, c0, false, false);
                c1 = __builtin_amdgcn_wmma_f32_16x16x32_f16(false, ah, false, bh1,
                                                            (short)0, c1, false, false);
                ah = an; bh0 = bn0; bh1 = bn1;
            }
        }
        // C/D layout: VGPR r: lanes 0-15 -> M=r, N=lane ; lanes 16-31 -> M=8+r
        {
            int n0 = wave * 32 + (lane & 15);
            int mb = (lane >> 4) << 3;
#pragma unroll
            for (int r = 0; r < 8; ++r) {
                int M = mb + r;
                lds_gates[M * NG + n0]      = c0[r];
                lds_gates[M * NG + n0 + 16] = c1[r];
            }
        }
        __syncthreads();

        // ---- stage 3: LSTM cell update + output ----
#pragma unroll
        for (int rr = 0; rr < 4; ++rr) {
            int idx = tid + 512 * rr;        // 2048 = 16 batches * 128 units
            int mm = idx >> 7;
            int j  = idx & 127;
            float gi = lds_gates[mm * NG + j];
            float gf = lds_gates[mm * NG + 128 + j];
            float gg = lds_gates[mm * NG + 256 + j];
            float go = lds_gates[mm * NG + 384 + j];
            float cold = lds_cf[mm * H_ + j];
            float cn = sigf(gf) * cold + sigf(gi) * tanhg(gg);
            float hn = sigf(go) * tanhg(cn);
            lds_cf[mm * H_ + j] = cn;
            lds_hf[mm * H_ + j] = hn;
            // pack h into A-fragment layout for next step's W_hh matmul
            int kt = j >> 5, kk = j & 31;
            int ln = ((kk >> 3) & 1) ? (16 + mm) : mm;
            int jj = (kk & 7) + ((kk & 16) ? 8 : 0);
            lds_h[(kt * 32 + ln) * 16 + jj] = (_Float16)hn;
            // out shape (B, T, H)
            out[((size_t)(btile * 16 + mm) * T_ + t) * H_ + j] = hn;
        }
        __syncthreads();
    }
}

// ---------------------------------------------------------------------------
extern "C" void kernel_launch(void* const* d_in, const int* in_sizes, int n_in,
                              void* d_out, int out_size, void* d_ws, size_t ws_size,
                              hipStream_t stream) {
    const float* input  = (const float*)d_in[0];
    const float* w_attn = (const float*)d_in[1];
    const float* b_attn = (const float*)d_in[2];
    const float* W_ih   = (const float*)d_in[3];
    const float* W_hh   = (const float*)d_in[4];
    const float* b_ih   = (const float*)d_in[5];
    const float* b_hh   = (const float*)d_in[6];
    float* out = (float*)d_out;

    _Float16* wih  = (_Float16*)d_ws;
    _Float16* whh  = wih + WIH_HALVES;
    float*    exw  = (float*)(whh + WHH_HALVES);
    float*    bias = exw + EX_FLOATS;

    {
        int total = WIH_HALVES + WHH_HALVES + BIAS_FLOATS;
        int blocks = (total + 255) / 256;
        enc_pack_kernel<<<blocks, 256, 0, stream>>>(W_ih, W_hh, b_ih, b_hh,
                                                    wih, whh, bias);
    }
    {
        int blocks = (B_ * D_ * 32) / 256;  // one wave per (b,d)
        enc_ex_kernel<<<blocks, 256, 0, stream>>>(input, w_attn, exw);
    }
    enc_main_kernel<<<16, 512, 0, stream>>>(input, w_attn, b_attn,
                                            wih, whh, exw, bias, out);
}
